// CLS_AttentionBlock_26036091748980
// MI455X (gfx1250) — matrix operand order repaired and checked by reading, not compiled, not run on hardware
//
#include <hip/hip_runtime.h>
#include <hip/hip_bf16.h>

typedef _Float16 half_t;
typedef __attribute__((ext_vector_type(16))) _Float16 v16h;
typedef __attribute__((ext_vector_type(8)))  _Float16 v8h;
typedef __attribute__((ext_vector_type(8)))  float    v8f;
typedef __attribute__((ext_vector_type(4)))  int      v4i;

#define C_DIM 512
#define HW    1024
#define NH    8
#define DH    64
#define NB    16
#define SLD   1041                      // f32 LDS row stride for score tile
#define BLD   72                        // half LDS row stride for B panels (16B-aligned rows)
#define SCALE 0.04419417382415922f      // 512^-0.5 (ref scales by full C, not dh)

// --- CDNA5 async global->LDS path (probe-guarded; fallback = synchronous copy)
#if defined(__has_builtin)
#if __has_builtin(__builtin_amdgcn_global_load_async_to_lds_b128) && \
    __has_builtin(__builtin_amdgcn_s_wait_asynccnt)
#define HAVE_ASYNC_LDS 1
#endif
#endif
#ifndef HAVE_ASYNC_LDS
#define HAVE_ASYNC_LDS 0
#endif

typedef __attribute__((address_space(1))) v4i gv4i_t;   // global int4
typedef __attribute__((address_space(3))) v4i lv4i_t;   // LDS int4

// copy 16 bytes (8 halves) global -> LDS, per lane
__device__ __forceinline__ void copy16_to_lds(const half_t* g, half_t* l) {
#if HAVE_ASYNC_LDS
  half_t* gm = const_cast<half_t*>(g);
  __builtin_amdgcn_global_load_async_to_lds_b128((gv4i_t*)gm, (lv4i_t*)l, 0, 0);
#else
  *(v8h*)l = *(const v8h*)g;
#endif
}

__device__ __forceinline__ void async_wait0() {
#if HAVE_ASYNC_LDS
  __builtin_amdgcn_s_wait_asynccnt(0);
#endif
}

__device__ __forceinline__ v8f wmma16(v16h a, v16h b, v8f c) {
  return __builtin_amdgcn_wmma_f32_16x16x32_f16(false, a, false, b, (short)0, c,
                                                false, false);
}

// A fragment (16x32 f16), row-major source, leading dim lda.
__device__ __forceinline__ v16h load_a16(const half_t* A, int lda, int lane) {
  const half_t* p = A + (lane & 15) * lda + ((lane >> 4) << 3);
  v8h lo = *(const v8h*)p;
  v8h hi = *(const v8h*)(p + 16);
  v16h f;
#pragma unroll
  for (int i = 0; i < 8; ++i) { f[i] = lo[i]; f[i + 8] = hi[i]; }
  return f;
}

// A fragment from an f32 LDS tile (convert to f16 per element).
__device__ __forceinline__ v16h load_a32(const float* A, int lda, int lane) {
  const float* p = A + (lane & 15) * lda + ((lane >> 4) << 3);
  v16h f;
#pragma unroll
  for (int i = 0; i < 8; ++i) {
    f[i]     = (half_t)p[i];
    f[i + 8] = (half_t)p[i + 16];
  }
  return f;
}

// B fragment (32x16 f16), row-major [K][N] source: lane%16 = N column, elements over K (table order).
__device__ __forceinline__ v16h load_b16(const half_t* B, int ldb, int lane) {
  const half_t* p = B + (size_t)((lane >> 4) << 3) * ldb + (lane & 15);
  v16h f;
#pragma unroll
  for (int i = 0; i < 8; ++i) {
    f[i]     = p[(size_t)i * ldb];
    f[i + 8] = p[(size_t)(i + 16) * ldb];
  }
  return f;
}

// ---------------------------------------------------------------------------
// Kernel 1: convert weights f32 -> f16, transposed to [K][N]. slot 0=wq 1=wk 2=wv 3=wo
__global__ void k_wconv(const float* __restrict__ wq, const float* __restrict__ wk,
                        const float* __restrict__ wv, const float* __restrict__ wo,
                        half_t* __restrict__ dst) {
  const float* W = (blockIdx.y == 0) ? wq : (blockIdx.y == 1) ? wk
                  : (blockIdx.y == 2) ? wv : wo;
  half_t* T = dst + (size_t)blockIdx.y * C_DIM * C_DIM;
  int i = blockIdx.x * blockDim.x + threadIdx.x;
  int k = i >> 9, n = i & 511;
  T[i] = (half_t)W[(size_t)n * C_DIM + k];       // T[k][n] = W[n][k]
}

// ---------------------------------------------------------------------------
// Kernel 2: GroupNorm, NCHW -> t[b][s][c] f16. One block per (batch, group).
__global__ void k_gn(const float* __restrict__ x, const float* __restrict__ gw,
                     const float* __restrict__ gb, half_t* __restrict__ t) {
  int b = blockIdx.x >> 5, g = blockIdx.x & 31;
  const float* xb = x + ((size_t)b * C_DIM + g * 16) * HW;
  int tid = threadIdx.x;
  float s = 0.f, s2 = 0.f;
  for (int i = tid; i < 16 * HW; i += 256) {
    float v = xb[i];
    s += v; s2 += v * v;
  }
  __shared__ float r1[256], r2[256];
  r1[tid] = s; r2[tid] = s2;
  __syncthreads();
  for (int off = 128; off > 0; off >>= 1) {
    if (tid < off) { r1[tid] += r1[tid + off]; r2[tid] += r2[tid + off]; }
    __syncthreads();
  }
  float mean = r1[0] * (1.f / 16384.f);
  float var  = r2[0] * (1.f / 16384.f) - mean * mean;
  float inv  = rsqrtf(var + 1e-5f);
  for (int i = tid; i < 16 * HW; i += 256) {
    int cc = i >> 10, sp = i & 1023;
    int c  = g * 16 + cc;
    float v = (xb[i] - mean) * inv * gw[c] + gb[c];
    t[((size_t)b * HW + sp) * C_DIM + c] = (half_t)v;
  }
}

// ---------------------------------------------------------------------------
// Kernel 3: QKV GEMM with double-buffered async-staged B panel in LDS.
// grid.z: 0=Q 1=K 2=V.  Q,V stored [b][h][s][dh]; K stored [b][h][dh][s].
__global__ void k_qkv(const half_t* __restrict__ t, const half_t* __restrict__ wT3,
                      const float* __restrict__ bq, const float* __restrict__ bk,
                      const float* __restrict__ bv,
                      half_t* __restrict__ Q, half_t* __restrict__ Kt,
                      half_t* __restrict__ V) {
  __shared__ half_t Bs[2][32 * BLD];
  int mode = blockIdx.z;
  const half_t* wT = wT3 + (size_t)mode * C_DIM * C_DIM;
  const float* bias = (mode == 0) ? bq : (mode == 1) ? bk : bv;
  int b  = blockIdx.y;
  int mt = blockIdx.x >> 3, nt = blockIdx.x & 7;   // 128x64 block tile
  int tid = threadIdx.x;
  int w  = tid >> 5, lane = tid & 31;
  int m0 = mt * 128 + w * 16;
  int n0 = nt * 64;
  const half_t* Abase = t + ((size_t)b * HW + m0) * C_DIM;

  // staging map: each thread owns 8 halves of the 32x64 panel
  int srow = tid >> 3, scol = (tid & 7) * 8;
  copy16_to_lds(wT + (size_t)srow * C_DIM + n0 + scol, &Bs[0][srow * BLD + scol]);

  v8f acc[4] = {};
  for (int ks = 0; ks < 16; ++ks) {
    int k0 = ks * 32;
    async_wait0();
    __syncthreads();
    if (ks + 1 < 16)   // overlap next panel's copy with this step's WMMAs
      copy16_to_lds(wT + (size_t)(k0 + 32 + srow) * C_DIM + n0 + scol,
                    &Bs[(ks + 1) & 1][srow * BLD + scol]);
    __builtin_prefetch(Abase + k0 + 32 + (lane & 15) * C_DIM, 0, 1);
    v16h a = load_a16(Abase + k0, C_DIM, lane);
    const half_t* Bb = Bs[ks & 1];
#pragma unroll
    for (int j = 0; j < 4; ++j) {
      v16h bf = load_b16(Bb + j * 16, BLD, lane);
      acc[j] = wmma16(a, bf, acc[j]);
    }
  }

  int hi8 = (lane >> 4) << 3;
#pragma unroll
  for (int j = 0; j < 4; ++j) {
    int col = n0 + j * 16 + (lane & 15);
    float bs = bias[col];
    int h = col >> 6, cw = col & 63;
#pragma unroll
    for (int r = 0; r < 8; ++r) {
      int s = m0 + r + hi8;
      float v = acc[j][r] + bs;
      if (mode == 1)
        Kt[(((size_t)b * NH + h) * DH + cw) * HW + s] = (half_t)v;
      else {
        half_t* dst = (mode == 0) ? Q : V;
        dst[(((size_t)b * NH + h) * HW + s) * DH + cw] = (half_t)v;
      }
    }
  }
}

// ---------------------------------------------------------------------------
// Kernel 4: attention for one (batch, head, 32-row q-tile). 8 waves.
__global__ void k_attn(const half_t* __restrict__ Q, const half_t* __restrict__ Kt,
                       const half_t* __restrict__ V, half_t* __restrict__ attn) {
  extern __shared__ char smem[];
  float*  S  = (float*)smem;                                       // [32][SLD]
  half_t* Qs = (half_t*)(smem + (size_t)32 * SLD * sizeof(float)); // [32][BLD]
  int q0 = blockIdx.x * 32;
  int h = blockIdx.y, b = blockIdx.z;
  int tid = threadIdx.x;
  int w = tid >> 5, lane = tid & 31;
  size_t bh = (size_t)b * NH + h;

  // async-stage the 32x64 Q tile (padded rows), one b128 per thread
  {
    const half_t* Qg = Q + (bh * HW + q0) * DH;
    int row = tid >> 3, col = (tid & 7) * 8;
    copy16_to_lds(Qg + row * DH + col, Qs + row * BLD + col);
    async_wait0();
  }
  __syncthreads();

  // S = SCALE * Q @ K^T   (M=32, N=1024, K=64)
  {
    const half_t* Kb = Kt + bh * DH * HW;
    int mt = w & 1;
    const half_t* QsA = Qs + mt * 16 * BLD;
    int hi8 = (lane >> 4) << 3;
    for (int nt = (w >> 1); nt < 64; nt += 4) {
      int j0 = nt * 16;
      v8f acc = {};
#pragma unroll
      for (int k0 = 0; k0 < DH; k0 += 32) {
        v16h a  = load_a16(QsA + k0, BLD, lane);
        v16h bf = load_b16(Kb + (size_t)k0 * HW + j0, HW, lane);
        acc = wmma16(a, bf, acc);
      }
      float* Sp = S + (size_t)(mt * 16) * SLD + j0 + (lane & 15);
#pragma unroll
      for (int r = 0; r < 8; ++r) Sp[(size_t)(r + hi8) * SLD] = acc[r] * SCALE;
    }
  }
  __syncthreads();

  // row softmax: 8 lanes per row, shuffle reductions within groups of 8
  {
    int row = tid >> 3, sub = tid & 7;
    float* Sr = S + (size_t)row * SLD;
    float mx = -3.4e38f;
    for (int j = sub; j < HW; j += 8) mx = fmaxf(mx, Sr[j]);
#pragma unroll
    for (int m = 1; m < 8; m <<= 1) mx = fmaxf(mx, __shfl_xor(mx, m, 8));
    float sm = 0.f;
    for (int j = sub; j < HW; j += 8) {
      float e = __expf(Sr[j] - mx);
      Sr[j] = e;
      sm += e;
    }
#pragma unroll
    for (int m = 1; m < 8; m <<= 1) sm += __shfl_xor(sm, m, 8);
    float inv = 1.0f / sm;
    for (int j = sub; j < HW; j += 8) Sr[j] *= inv;
  }
  __syncthreads();

  // O = P @ V   (M=32, N=64, K=1024); one 16x16 tile per wave
  {
    int mt2 = w & 1, nt2 = (w >> 1) & 3;
    const half_t* Vb = V + bh * HW * DH + nt2 * 16;
    const float*  SA = S + (size_t)(mt2 * 16) * SLD;
    v8f acc = {};
    for (int k0 = 0; k0 < HW; k0 += 32) {
      v16h a  = load_a32(SA + k0, SLD, lane);
      v16h bf = load_b16(Vb + (size_t)k0 * DH, DH, lane);
      acc = wmma16(a, bf, acc);
    }
    int hi8 = (lane >> 4) << 3;
    int col = h * DH + nt2 * 16 + (lane & 15);
#pragma unroll
    for (int r = 0; r < 8; ++r) {
      int s = q0 + mt2 * 16 + r + hi8;
      attn[((size_t)b * HW + s) * C_DIM + col] = (half_t)acc[r];
    }
  }
}

// ---------------------------------------------------------------------------
// Kernel 5: out = attn @ Wo.T (async-staged B panel), NCHW transpose, +residual.
__global__ void k_out(const half_t* __restrict__ attn, const half_t* __restrict__ woT,
                      const float* __restrict__ x, float* __restrict__ out) {
  __shared__ half_t Bs[2][32 * BLD];
  int b  = blockIdx.y;
  int mt = blockIdx.x >> 3, nt = blockIdx.x & 7;
  int tid = threadIdx.x;
  int w  = tid >> 5, lane = tid & 31;
  int m0 = mt * 128 + w * 16;
  int n0 = nt * 64;
  const half_t* Abase = attn + ((size_t)b * HW + m0) * C_DIM;

  int srow = tid >> 3, scol = (tid & 7) * 8;
  copy16_to_lds(woT + (size_t)srow * C_DIM + n0 + scol, &Bs[0][srow * BLD + scol]);

  v8f acc[4] = {};
  for (int ks = 0; ks < 16; ++ks) {
    int k0 = ks * 32;
    async_wait0();
    __syncthreads();
    if (ks + 1 < 16)
      copy16_to_lds(woT + (size_t)(k0 + 32 + srow) * C_DIM + n0 + scol,
                    &Bs[(ks + 1) & 1][srow * BLD + scol]);
    __builtin_prefetch(Abase + k0 + 32 + (lane & 15) * C_DIM, 0, 1);
    v16h a = load_a16(Abase + k0, C_DIM, lane);
    const half_t* Bb = Bs[ks & 1];
#pragma unroll
    for (int j = 0; j < 4; ++j) {
      v16h bf = load_b16(Bb + j * 16, BLD, lane);
      acc[j] = wmma16(a, bf, acc[j]);
    }
  }

  int hi8 = (lane >> 4) << 3;
#pragma unroll
  for (int j = 0; j < 4; ++j) {
    int c = n0 + j * 16 + (lane & 15);
#pragma unroll
    for (int r = 0; r < 8; ++r) {
      int s = m0 + r + hi8;
      size_t idx = ((size_t)b * C_DIM + c) * HW + s;
      out[idx] = acc[j][r] + x[idx];
    }
  }
}

// ---------------------------------------------------------------------------
extern "C" void kernel_launch(void* const* d_in, const int* in_sizes, int n_in,
                              void* d_out, int out_size, void* d_ws, size_t ws_size,
                              hipStream_t stream) {
  (void)in_sizes; (void)n_in; (void)out_size; (void)ws_size;
  const float* x  = (const float*)d_in[0];
  const float* gw = (const float*)d_in[1];
  const float* gb = (const float*)d_in[2];
  const float* wq = (const float*)d_in[3];
  const float* bq = (const float*)d_in[4];
  const float* wk = (const float*)d_in[5];
  const float* bk = (const float*)d_in[6];
  const float* wv = (const float*)d_in[7];
  const float* bv = (const float*)d_in[8];
  const float* wo = (const float*)d_in[9];
  float* out = (float*)d_out;

  const size_t TSZ = (size_t)NB * HW * C_DIM;   // 8,388,608 halves per tensor
  half_t* t    = (half_t*)d_ws;                 // GN output [b][s][c]
  half_t* Q    = t    + TSZ;                    // [b][h][s][dh]
  half_t* Kt   = Q    + TSZ;                    // [b][h][dh][s]
  half_t* V    = Kt   + TSZ;                    // [b][h][s][dh]
  half_t* attn = V    + TSZ;                    // [b][s][c]
  half_t* wT   = attn + TSZ;                    // 4x [512][512] f16 (q,k,v,o)

  k_wconv<<<dim3(1024, 4), 256, 0, stream>>>(wq, wk, wv, wo, wT);
  k_gn   <<<dim3(NB * 32), 256, 0, stream>>>(x, gw, gb, t);
  k_qkv  <<<dim3(64, NB, 3), 256, 0, stream>>>(t, wT, bq, bk, bv, Q, Kt, V);

  size_t smem = (size_t)32 * SLD * sizeof(float) + (size_t)32 * BLD * sizeof(half_t);
  (void)hipFuncSetAttribute(reinterpret_cast<const void*>(k_attn),
                            hipFuncAttributeMaxDynamicSharedMemorySize, (int)smem);
  k_attn <<<dim3(HW / 32, NH, NB), 256, smem, stream>>>(Q, Kt, V, attn);

  k_out  <<<dim3(64, NB), 256, 0, stream>>>(attn, wT + 3 * (size_t)C_DIM * C_DIM, x, out);
}
